// EnergyAwareAdaptiveFusion_48490180771932
// MI455X (gfx1250) — compile-verified, matrix operand order checked by reading, not run on hardware
//
#include <hip/hip_runtime.h>
#include <stdint.h>

// EnergyAwareAdaptiveFusion for MI455X (gfx1250, wave32, WMMA bf16 16x16x32).
// GEMM waves own a 32x64 output tile (2 m-tiles x 4 n-tiles) so each
// k-iteration issues 8 v_wmma against 12 b128 loads (W frags amortized
// over 2 m-tiles; 8 waves/block share one n-chunk -> W hot in WGP$/L2).

#define B_TOK 16384
#define DIM   1024
#define NH    16
#define HDIM  64
#define LN_EPS 1e-5f

typedef __bf16 bf16_t;
typedef __attribute__((ext_vector_type(16))) __bf16 v16bf;
typedef __attribute__((ext_vector_type(8)))  float  v8f;

// ---------------- bf16 helpers (RNE) ----------------
__device__ __forceinline__ float bf2f(bf16_t x) {
  uint16_t b = __builtin_bit_cast(uint16_t, x);
  return __uint_as_float(((uint32_t)b) << 16);
}
__device__ __forceinline__ bf16_t f2bf(float f) {
  uint32_t u = __float_as_uint(f);
  u = (u + 0x7FFFu + ((u >> 16) & 1u)) >> 16;
  return __builtin_bit_cast(bf16_t, (uint16_t)u);
}
__device__ __forceinline__ uint32_t pack_bf2(float x, float y) {
  uint32_t a = __float_as_uint(x), b = __float_as_uint(y);
  a = (a + 0x7FFFu + ((a >> 16) & 1u)) >> 16;
  b = (b + 0x7FFFu + ((b >> 16) & 1u)) >> 16;
  return (a & 0xFFFFu) | (b << 16);
}

// ---------------- WMMA fragment loads ----------------
// A (16x32 bf16, row-major source, stride in elements):
// lane l holds row m0+(l&15); half h=l>>4 selects k = {h*8..h*8+7, 16+h*8..16+h*8+7}
// -> two contiguous 16B loads per lane.
__device__ __forceinline__ v16bf ldA(const bf16_t* A, int stride, int m, int k0, int lane) {
  const bf16_t* p = A + (size_t)m * stride + k0 + ((lane >> 4) << 3);
  union { v16bf v; uint4 q[2]; } u;
  u.q[0] = *(const uint4*)p;
  u.q[1] = *(const uint4*)(p + 16);
  return u.v;
}
// W packed layout: u32 Wpk[((nt*KT + kt)*32 + lane)*8 + j] -> 32B contiguous per lane.
__device__ __forceinline__ v16bf ldW(const uint32_t* Wpk, int KT, int nt, int kt, int lane) {
  const uint32_t* p = Wpk + ((((size_t)nt * KT + kt) << 5) + lane) * 8;
  union { v16bf v; uint4 q[2]; } u;
  u.q[0] = *(const uint4*)p;
  u.q[1] = *(const uint4*)(p + 4);
  return u.v;
}

// Accumulate a 32x64 output tile (2 m-tiles x 4 n-tiles) over nkt k-tiles of 32.
__device__ __forceinline__ void wmma_accum_2x4(const bf16_t* A, int stride,
    const uint32_t* Wpk, int KT, int nt0, int ktW0, int nkt,
    int m0, int lane, v8f acc[2][4]) {
  int mA = m0 + (lane & 15);
  for (int i = 0; i < nkt; ++i) {
    v16bf a0 = ldA(A, stride, mA,      i << 5, lane);
    v16bf a1 = ldA(A, stride, mA + 16, i << 5, lane);
#pragma unroll
    for (int t = 0; t < 4; ++t) {
      v16bf b = ldW(Wpk, KT, nt0 + t, ktW0 + i, lane);
      acc[0][t] = __builtin_amdgcn_wmma_f32_16x16x32_bf16(
          false, a0, false, b, (short)0, acc[0][t], false, false);
      acc[1][t] = __builtin_amdgcn_wmma_f32_16x16x32_bf16(
          false, a1, false, b, (short)0, acc[1][t], false, false);
    }
  }
}

// C/D layout: VGPR r -> m = m0 + u*16 + r + (lane>=16)*8 ; n = n0 + t*16 + (lane&15)
#define EPILOGUE_BEGIN(acc, m0, n0, lane)                               \
  _Pragma("unroll") for (int u = 0; u < 2; ++u) {                       \
    _Pragma("unroll") for (int t = 0; t < 4; ++t) {                     \
      _Pragma("unroll") for (int r = 0; r < 8; ++r) {                   \
        int em = (m0) + u * 16 + r + (((lane) >> 4) << 3);              \
        int en = (n0) + t * 16 + ((lane) & 15);                         \
        float ev = acc[u][t][r];
#define EPILOGUE_END } } }

// Decompose global wave id into a 32-row m-block and 64-col n-chunk.
#define GEMM_PROLOGUE                                                   \
  int lane = threadIdx.x & 31;                                          \
  int gw = blockIdx.x * 8 + (threadIdx.x >> 5);                         \
  const int MB = B_TOK / 32;                                            \
  int m0 = (gw % MB) * 32, n0 = (gw / MB) * 64;                         \
  v8f acc[2][4] = {};

// ---------------- prep kernels ----------------
__global__ __launch_bounds__(256) void cvt_bf16_kernel(const float* __restrict__ X,
                                                       uint32_t* __restrict__ Y, size_t n2) {
  size_t i = (size_t)blockIdx.x * 256 + threadIdx.x;
  if (i >= n2) return;
  Y[i] = pack_bf2(X[2 * i], X[2 * i + 1]);
}

// Pack row-major fp32 W[K][N] into the WMMA B-fragment bf16 layout.
__global__ __launch_bounds__(256) void pack_w_kernel(const float* __restrict__ W,
                                                     uint32_t* __restrict__ Wpk,
                                                     int K, int N) {
  size_t idx = (size_t)blockIdx.x * 256 + threadIdx.x;
  size_t total = (size_t)K * N / 2;
  if (idx >= total) return;
  int j    = idx & 7;
  int lane = (idx >> 3) & 31;
  int KT   = K >> 5;
  int kt   = (int)((idx >> 8) % KT);
  int nt   = (int)((idx >> 8) / KT);
  int h    = lane >> 4;
  int n    = nt * 16 + (lane & 15);
  int kin  = (j < 4) ? (h * 8 + 2 * j) : (16 + h * 8 + 2 * (j - 4));
  int k    = kt * 32 + kin;
  Wpk[idx] = pack_bf2(W[(size_t)k * N + n], W[(size_t)(k + 1) * N + n]);
}

// ---------------- phase kernels ----------------
// gate = sigmoid(img@Wg_top + txt@Wg_bot + bg); fused0 = gate*img + (1-gate)*txt
__global__ __launch_bounds__(256) void gate_fuse_kernel(
    const bf16_t* __restrict__ imgb, const bf16_t* __restrict__ txtb,
    const uint32_t* __restrict__ WgPk, const float* __restrict__ bg,
    const float* __restrict__ img, const float* __restrict__ txt,
    float* __restrict__ fused0) {
  GEMM_PROLOGUE
  wmma_accum_2x4(imgb, DIM, WgPk, 64, n0 >> 4, 0,  32, m0, lane, acc);
  wmma_accum_2x4(txtb, DIM, WgPk, 64, n0 >> 4, 32, 32, m0, lane, acc);
  EPILOGUE_BEGIN(acc, m0, n0, lane)
    float g = 1.0f / (1.0f + expf(-(ev + bg[en])));
    size_t o = (size_t)em * DIM + en;
    fused0[o] = g * img[o] + (1.0f - g) * txt[o];
  EPILOGUE_END
}

// qkv_s = x_s @ Wqkv + bqkv  (bf16 out), grid.z = stream
__global__ __launch_bounds__(256) void qkv_kernel(
    const bf16_t* __restrict__ imgb, const bf16_t* __restrict__ txtb,
    const uint32_t* __restrict__ WqkvPk, const float* __restrict__ bqkv,
    bf16_t* __restrict__ qkv0, bf16_t* __restrict__ qkv1) {
  const bf16_t* A  = blockIdx.z ? txtb : imgb;
  bf16_t*       Co = blockIdx.z ? qkv1 : qkv0;
  GEMM_PROLOGUE
  wmma_accum_2x4(A, DIM, WqkvPk, 32, n0 >> 4, 0, 32, m0, lane, acc);
  EPILOGUE_BEGIN(acc, m0, n0, lane)
    Co[(size_t)em * (3 * DIM) + en] = f2bf(ev + bqkv[en]);
  EPILOGUE_END
}

// Sequence-length-2 attention; fold mean over the two query rows:
// ctx_mean = 0.5*(a00+a10)*v_img + 0.5*(a01+a11)*v_txt   (per head)
__global__ __launch_bounds__(256) void attn_kernel(
    const bf16_t* __restrict__ qkv0, const bf16_t* __restrict__ qkv1,
    bf16_t* __restrict__ ctxm) {
  int idx = blockIdx.x * 256 + threadIdx.x;   // B*NH threads
  int b = idx >> 4, h = idx & 15;
  const bf16_t* b0 = qkv0 + (size_t)b * 3 * DIM + h * HDIM;
  const bf16_t* b1 = qkv1 + (size_t)b * 3 * DIM + h * HDIM;
  const bf16_t *qi = b0, *qt = b1, *ki = b0 + DIM, *kt = b1 + DIM;
  const bf16_t *vi = b0 + 2 * DIM, *vt = b1 + 2 * DIM;
  float s00 = 0, s01 = 0, s10 = 0, s11 = 0;
#pragma unroll 8
  for (int d = 0; d < HDIM; ++d) {
    float a = bf2f(qi[d]), c = bf2f(qt[d]), e = bf2f(ki[d]), f = bf2f(kt[d]);
    s00 += a * e; s01 += a * f; s10 += c * e; s11 += c * f;
  }
  const float sc = 0.125f;  // 1/sqrt(64)
  s00 *= sc; s01 *= sc; s10 *= sc; s11 *= sc;
  float mx0 = fmaxf(s00, s01), e00 = expf(s00 - mx0), e01 = expf(s01 - mx0);
  float r0 = 1.0f / (e00 + e01);
  float mx1 = fmaxf(s10, s11), e10 = expf(s10 - mx1), e11 = expf(s11 - mx1);
  float r1 = 1.0f / (e10 + e11);
  float ci = 0.5f * (e00 * r0 + e10 * r1);
  float ct = 0.5f * (e01 * r0 + e11 * r1);
  bf16_t* o = ctxm + (size_t)b * DIM + h * HDIM;
#pragma unroll 8
  for (int d = 0; d < HDIM; ++d) o[d] = f2bf(ci * bf2f(vi[d]) + ct * bf2f(vt[d]));
}

// y = fused0 + ctx_mean @ Wo + bo  (in place on fused0)
__global__ __launch_bounds__(256) void attnout_kernel(
    const bf16_t* __restrict__ ctxb, const uint32_t* __restrict__ WoPk,
    const float* __restrict__ bo, float* __restrict__ y) {
  GEMM_PROLOGUE
  wmma_accum_2x4(ctxb, DIM, WoPk, 32, n0 >> 4, 0, 32, m0, lane, acc);
  EPILOGUE_BEGIN(acc, m0, n0, lane)
    size_t o = (size_t)em * DIM + en;
    y[o] = y[o] + ev + bo[en];
  EPILOGUE_END
}

__global__ __launch_bounds__(256) void ln_kernel(
    const float* __restrict__ y, const float* __restrict__ g,
    const float* __restrict__ b, bf16_t* __restrict__ o) {
  __shared__ float s1[256], s2[256];
  int row = blockIdx.x, tid = threadIdx.x;
  const float* yr = y + (size_t)row * DIM;
  float x[4], s = 0, q = 0;
#pragma unroll
  for (int j = 0; j < 4; ++j) { x[j] = yr[tid + j * 256]; s += x[j]; q += x[j] * x[j]; }
  s1[tid] = s; s2[tid] = q; __syncthreads();
  for (int st = 128; st > 0; st >>= 1) {
    if (tid < st) { s1[tid] += s1[tid + st]; s2[tid] += s2[tid + st]; }
    __syncthreads();
  }
  float mu  = s1[0] * (1.0f / DIM);
  float var = s2[0] * (1.0f / DIM) - mu * mu;
  float inv = rsqrtf(var + LN_EPS);
  bf16_t* orow = o + (size_t)row * DIM;
#pragma unroll
  for (int j = 0; j < 4; ++j) {
    int i = tid + j * 256;
    orow[i] = f2bf((x[j] - mu) * inv * g[i] + b[i]);
  }
}

// h1 = gelu_exact(ln @ Wf1 + bf1)
__global__ __launch_bounds__(256) void ffn1_kernel(
    const bf16_t* __restrict__ lnb, const uint32_t* __restrict__ Wf1Pk,
    const float* __restrict__ bf1, bf16_t* __restrict__ h1) {
  GEMM_PROLOGUE
  wmma_accum_2x4(lnb, DIM, Wf1Pk, 32, n0 >> 4, 0, 32, m0, lane, acc);
  EPILOGUE_BEGIN(acc, m0, n0, lane)
    float xv = ev + bf1[en];
    float gv = 0.5f * xv * (1.0f + erff(xv * 0.70710678118f));
    h1[(size_t)em * (4 * DIM) + en] = f2bf(gv);
  EPILOGUE_END
}

// out = rc==0 ? img : rc==1 ? txt : (h1 @ Wf2 + bf2)
__global__ __launch_bounds__(256) void ffn2_route_kernel(
    const bf16_t* __restrict__ h1, const uint32_t* __restrict__ Wf2Pk,
    const float* __restrict__ bf2, const int* __restrict__ route,
    const float* __restrict__ img, const float* __restrict__ txt,
    float* __restrict__ out) {
  GEMM_PROLOGUE
  wmma_accum_2x4(h1, 4 * DIM, Wf2Pk, 128, n0 >> 4, 0, 128, m0, lane, acc);
  EPILOGUE_BEGIN(acc, m0, n0, lane)
    int rc = route[em];
    size_t o = (size_t)em * DIM + en;
    out[o] = (rc == 0) ? img[o] : (rc == 1) ? txt[o] : (ev + bf2[en]);
  EPILOGUE_END
}

// ---------------- launch ----------------
extern "C" void kernel_launch(void* const* d_in, const int* in_sizes, int n_in,
                              void* d_out, int out_size, void* d_ws, size_t ws_size,
                              hipStream_t stream) {
  const float* img   = (const float*)d_in[0];
  const float* txt   = (const float*)d_in[1];
  const int*   route = (const int*)d_in[2];
  const float* Wg    = (const float*)d_in[3];
  const float* bg    = (const float*)d_in[4];
  const float* Wqkv  = (const float*)d_in[5];
  const float* bqkv  = (const float*)d_in[6];
  const float* Wo    = (const float*)d_in[7];
  const float* bo    = (const float*)d_in[8];
  const float* gamma = (const float*)d_in[9];
  const float* beta  = (const float*)d_in[10];
  const float* Wf1   = (const float*)d_in[11];
  const float* bf1   = (const float*)d_in[12];
  const float* Wf2   = (const float*)d_in[13];
  const float* bf2   = (const float*)d_in[14];
  float* out = (float*)d_out;

  char* ws = (char*)d_ws;
  size_t off = 0;
  auto wsalloc = [&](size_t bytes) -> char* {
    char* p = ws + off;
    off += (bytes + 255) & ~(size_t)255;
    return p;
  };
  const size_t SB = (size_t)B_TOK;
  bf16_t*   imgb    = (bf16_t*)  wsalloc(SB * DIM * 2);
  bf16_t*   txtb    = (bf16_t*)  wsalloc(SB * DIM * 2);
  uint32_t* WgPk    = (uint32_t*)wsalloc((size_t)2 * DIM * DIM * 2);
  uint32_t* WqkvPk  = (uint32_t*)wsalloc((size_t)DIM * 3 * DIM * 2);
  uint32_t* WoPk    = (uint32_t*)wsalloc((size_t)DIM * DIM * 2);
  uint32_t* Wf1Pk   = (uint32_t*)wsalloc((size_t)DIM * 4 * DIM * 2);
  uint32_t* Wf2Pk   = (uint32_t*)wsalloc((size_t)4 * DIM * DIM * 2);
  float*    fused0  = (float*)   wsalloc(SB * DIM * 4);
  bf16_t*   lnb     = (bf16_t*)  wsalloc(SB * DIM * 2);
  bf16_t*   ctxb    = (bf16_t*)  wsalloc(SB * DIM * 2);
  bf16_t*   qkv0    = (bf16_t*)  wsalloc(SB * 3 * DIM * 2);
  bf16_t*   qkv1    = (bf16_t*)  wsalloc(SB * 3 * DIM * 2);
  bf16_t*   h1      = qkv0;  // alias: qkv dead after attention, h1 = B x 4D

  const int MB = B_TOK / 32;   // 32-row m-blocks per GEMM wave
  size_t n2 = SB * DIM / 2;

  cvt_bf16_kernel<<<(unsigned)((n2 + 255) / 256), 256, 0, stream>>>(img, (uint32_t*)imgb, n2);
  cvt_bf16_kernel<<<(unsigned)((n2 + 255) / 256), 256, 0, stream>>>(txt, (uint32_t*)txtb, n2);

  pack_w_kernel<<<(2 * DIM * DIM / 2) / 256, 256, 0, stream>>>(Wg,   WgPk,   2 * DIM, DIM);
  pack_w_kernel<<<(DIM * 3 * DIM / 2) / 256, 256, 0, stream>>>(Wqkv, WqkvPk, DIM, 3 * DIM);
  pack_w_kernel<<<(DIM * DIM / 2)     / 256, 256, 0, stream>>>(Wo,   WoPk,   DIM, DIM);
  pack_w_kernel<<<(DIM * 4 * DIM / 2) / 256, 256, 0, stream>>>(Wf1,  Wf1Pk,  DIM, 4 * DIM);
  pack_w_kernel<<<(4 * DIM * DIM / 2) / 256, 256, 0, stream>>>(Wf2,  Wf2Pk,  4 * DIM, DIM);

  gate_fuse_kernel<<<MB * (DIM / 64) / 8, 256, 0, stream>>>(imgb, txtb, WgPk, bg, img, txt, fused0);

  dim3 gq(MB * (3 * DIM / 64) / 8, 1, 2);
  qkv_kernel<<<gq, 256, 0, stream>>>(imgb, txtb, WqkvPk, bqkv, qkv0, qkv1);

  attn_kernel<<<(B_TOK * NH) / 256, 256, 0, stream>>>(qkv0, qkv1, ctxb);

  attnout_kernel<<<MB * (DIM / 64) / 8, 256, 0, stream>>>(ctxb, WoPk, bo, fused0);

  ln_kernel<<<B_TOK, 256, 0, stream>>>(fused0, gamma, beta, lnb);

  ffn1_kernel<<<MB * (4 * DIM / 64) / 8, 256, 0, stream>>>(lnb, Wf1Pk, bf1, h1);

  ffn2_route_kernel<<<MB * (DIM / 64) / 8, 256, 0, stream>>>(h1, Wf2Pk, bf2, route, img, txt, out);
}
